// Intra_graph_53025666236862
// MI455X (gfx1250) — matrix operand order repaired and checked by reading, not compiled, not run on hardware
//
#include <hip/hip_runtime.h>

typedef __bf16 bf16_t;
typedef __bf16 v8bf  __attribute__((ext_vector_type(8)));
typedef __bf16 v16bf __attribute__((ext_vector_type(16)));
typedef float  v8f   __attribute__((ext_vector_type(8)));

#define B_   4
#define C_   1024
#define WH_  4096
#define IN_  256
#define ND_  64
#define DC_  128

// ---------------------------------------------------------------------------
// Fragment loaders for v_wmma_f32_16x16x32_bf16 (CDNA5 ISA 7.12.2 layouts).
// A row-major [M][K]; B supplied transposed row-major BT[N][K] so both
// fragments are contiguous 16B chunks -> plain global_load_b128.
// ---------------------------------------------------------------------------
__device__ __forceinline__ void load_frags(const bf16_t* __restrict__ Ab, int lda,
                                           const bf16_t* __restrict__ Bb, int ldbt,
                                           int m_base, int n_base, int r, int half,
                                           int k0, v16bf a[2], v16bf bfv[2])
{
    #pragma unroll
    for (int t = 0; t < 2; ++t) {
        // A 16x32 fragment: lane r = row; K chunks [half*8,+8) and [16+half*8,+8)
        const bf16_t* pa = Ab + (long long)(m_base + t * 16 + r) * lda + (k0 + half * 8);
        v8bf alo = *(const v8bf*)(pa);
        v8bf ahi = *(const v8bf*)(pa + 16);
        a[t] = __builtin_shufflevector(alo, ahi, 0,1,2,3,4,5,6,7,8,9,10,11,12,13,14,15);
        // B 32x16 fragment from BT[N][K]: lane r = col; K chunks [half*8,+8) and [16+half*8,+8)
        const bf16_t* pb = Bb + (long long)(n_base + t * 16 + r) * ldbt + (k0 + half * 8);
        v8bf blo = *(const v8bf*)(pb);
        v8bf bhi = *(const v8bf*)(pb + 16);
        bfv[t] = __builtin_shufflevector(blo, bhi, 0,1,2,3,4,5,6,7,8,9,10,11,12,13,14,15);
    }
}

// ---------------------------------------------------------------------------
// Generic bf16 WMMA GEMM: D[M][N] = A[M][K] * BT[N][K]^T (+bias[m]).
// Block = 128 threads = 4 waves in 2x2, block tile 64x64, wave tile 32x32.
// Software-pipelined double buffer over K (K always a multiple of 64 here):
// fragment loads for step k+32 / k+64 are issued before the WMMAs consuming
// step k, so LOADcnt covers latency instead of stalling at s_wait_loadcnt 0.
// Epilogue: optional f32 store, bf16 store, bf16 transposed store.
// Grid: (N/64, M/64, batch). M,N multiples of 64, K multiple of 64.
// ---------------------------------------------------------------------------
__global__ __launch_bounds__(128)
void gemm_bf16_k(const bf16_t* __restrict__ A,  long long sA,  int lda,
                 const bf16_t* __restrict__ BT, long long sBT, int ldbt,
                 int K,
                 const float*  __restrict__ bias,
                 float*  __restrict__ Df,  long long sDf,  int lddf,
                 bf16_t* __restrict__ Db,  long long sDb,  int lddb,
                 bf16_t* __restrict__ DTb, long long sDTb, int lddtb)
{
    const int tid  = threadIdx.x;
    const int lane = tid & 31;
    const int w    = tid >> 5;
    const int r    = lane & 15;
    const int half = lane >> 4;
    const int z    = blockIdx.z;
    const int m_base = blockIdx.y * 64 + (w >> 1) * 32;
    const int n_base = blockIdx.x * 64 + (w & 1) * 32;

    const bf16_t* Ab = A  + (long long)z * sA;
    const bf16_t* Bb = BT + (long long)z * sBT;

    v8f zero8 = {0.f,0.f,0.f,0.f,0.f,0.f,0.f,0.f};
    v8f acc[2][2];
    #pragma unroll
    for (int i = 0; i < 2; ++i)
        #pragma unroll
        for (int j = 0; j < 2; ++j) acc[i][j] = zero8;

    v16bf a0[2], b0[2], a1[2], b1[2];
    load_frags(Ab, lda, Bb, ldbt, m_base, n_base, r, half, 0, a0, b0);

    for (int k0 = 0; k0 < K - 32; k0 += 64) {
        __builtin_prefetch(Ab + (long long)(m_base + r) * lda + (k0 + 64), 0, 1);
        __builtin_prefetch(Bb + (long long)(n_base + r) * ldbt + (k0 + 64), 0, 1);
        // issue loads for k0+32 before consuming k0
        load_frags(Ab, lda, Bb, ldbt, m_base, n_base, r, half, k0 + 32, a1, b1);
        #pragma unroll
        for (int i = 0; i < 2; ++i)
            #pragma unroll
            for (int j = 0; j < 2; ++j)
                acc[i][j] = __builtin_amdgcn_wmma_f32_16x16x32_bf16(
                    false, a0[i], false, b0[j], (short)0, acc[i][j], false, false);
        if (k0 + 64 < K)   // issue loads for k0+64 before consuming k0+32
            load_frags(Ab, lda, Bb, ldbt, m_base, n_base, r, half, k0 + 64, a0, b0);
        #pragma unroll
        for (int i = 0; i < 2; ++i)
            #pragma unroll
            for (int j = 0; j < 2; ++j)
                acc[i][j] = __builtin_amdgcn_wmma_f32_16x16x32_bf16(
                    false, a1[i], false, b1[j], (short)0, acc[i][j], false, false);
    }

    // C/D layout: VGPR v holds row m = v + half*8, col n = lane&15
    #pragma unroll
    for (int i = 0; i < 2; ++i) {
        #pragma unroll
        for (int j = 0; j < 2; ++j) {
            #pragma unroll
            for (int v = 0; v < 8; ++v) {
                int m = m_base + i * 16 + half * 8 + v;
                int n = n_base + j * 16 + r;
                float val = acc[i][j][v];
                if (bias) val += bias[m];
                if (Df)  Df [(long long)z * sDf  + (long long)m * lddf  + n] = val;
                if (Db)  Db [(long long)z * sDb  + (long long)m * lddb  + n] = (bf16_t)val;
                if (DTb) DTb[(long long)z * sDTb + (long long)n * lddtb + m] = (bf16_t)val;
            }
        }
    }
}

// ---- transpose + f32->bf16: x[b][1024][4096] -> xT[b][4096][1024] ----------
// LDS tile filled with CDNA5 async global->LDS loads (ASYNCcnt tracked);
// LDS offset = low 32 bits of the flat shared-aperture address (ISA 10.2).
__global__ __launch_bounds__(256)
void transpose_conv_k(const float* __restrict__ in, bf16_t* __restrict__ out)
{
    __shared__ float t[32][33];
    int p0 = blockIdx.x * 32, c0 = blockIdx.y * 32, b = blockIdx.z;
    const float* ib = in + (long long)b * C_ * WH_;
    bf16_t* ob = out + (long long)b * WH_ * C_;
    int tx = threadIdx.x & 31, ty = threadIdx.x >> 5;
    #pragma unroll
    for (int k = 0; k < 4; ++k) {
        unsigned long long ga =
            (unsigned long long)(ib + (long long)(c0 + ty + k * 8) * WH_ + p0 + tx);
        unsigned lds = (unsigned)(unsigned long long)&t[ty + k * 8][tx];
        asm volatile("global_load_async_to_lds_b32 %0, %1, off"
                     :: "v"(lds), "v"(ga) : "memory");
    }
    asm volatile("s_wait_asynccnt 0x0" ::: "memory");
    __syncthreads();
    #pragma unroll
    for (int k = 0; k < 4; ++k)
        ob[(long long)(p0 + ty + k * 8) * C_ + c0 + tx] = (bf16_t)t[tx][ty + k * 8];
}

__global__ void f2bf_k(const float* __restrict__ in, bf16_t* __restrict__ out, int n)
{
    int i = blockIdx.x * 256 + threadIdx.x;
    if (i < n) out[i] = (bf16_t)in[i];
}

// muT[b][n][c] = proto[c][n];  pi[b][n] = pi0[n]
__global__ void init_k(const float* __restrict__ proto, const float* __restrict__ pi0,
                       bf16_t* __restrict__ muT, float* __restrict__ pi)
{
    int i = blockIdx.x * 256 + threadIdx.x;         // B_*ND_*IN_ total
    int rest = i & (ND_ * IN_ - 1);
    int n = rest >> 8, c = rest & 255;
    muT[i] = (bf16_t)proto[c * ND_ + n];
    if (i < B_ * ND_) pi[i] = pi0[i & (ND_ - 1)];
}

__global__ void zero_k(float* p, int n)
{
    int i = blockIdx.x * 256 + threadIdx.x;
    if (i < n) p[i] = 0.f;
}

// ---- EM step 1: per-pixel softmax(lik)*pi, normalized over nodes -----------
__global__ __launch_bounds__(256)
void em_post_k(const float* __restrict__ likT, const float* __restrict__ pi,
               bf16_t* __restrict__ post_bf, float* __restrict__ colsum)
{
    int b = blockIdx.y;
    int m = blockIdx.x * 256 + threadIdx.x;
    const float* lt = likT + (long long)b * ND_ * WH_ + m;
    float e[ND_];
    float mx = -3.4e38f;
    #pragma unroll
    for (int n = 0; n < ND_; ++n) { e[n] = lt[n * WH_]; mx = fmaxf(mx, e[n]); }
    float s = 0.f;
    #pragma unroll
    for (int n = 0; n < ND_; ++n) { e[n] = __expf(e[n] - mx) * pi[b * ND_ + n]; s += e[n]; }
    float inv = 1.f / (1e-18f + s);

    __shared__ float cs[ND_];
    if (threadIdx.x < ND_) cs[threadIdx.x] = 0.f;
    __syncthreads();

    bf16_t* po = post_bf + ((long long)b * WH_ + m) * ND_;
    int lane = threadIdx.x & 31;
    #pragma unroll
    for (int n = 0; n < ND_; ++n) {
        float p = e[n] * inv;
        po[n] = (bf16_t)p;
        float ps = p;                       // wave32 tree reduction
        ps += __shfl_down(ps, 16);
        ps += __shfl_down(ps, 8);
        ps += __shfl_down(ps, 4);
        ps += __shfl_down(ps, 2);
        ps += __shfl_down(ps, 1);
        if (lane == 0) atomicAdd(&cs[n], ps);
    }
    __syncthreads();
    if (threadIdx.x < ND_) atomicAdd(&colsum[b * ND_ + threadIdx.x], cs[threadIdx.x]);
}

// ---- EM step 2: prob = post/colsum (stored transposed), pi = colsum/wh -----
__global__ __launch_bounds__(256)
void em_prob_k(const bf16_t* __restrict__ post_bf, const float* __restrict__ colsum,
               bf16_t* __restrict__ probT, float* __restrict__ pi)
{
    int b = blockIdx.y;
    int m = blockIdx.x * 256 + threadIdx.x;
    __shared__ float inv[ND_];
    if (threadIdx.x < ND_) inv[threadIdx.x] = 1.f / (1e-18f + colsum[b * ND_ + threadIdx.x]);
    __syncthreads();
    const bf16_t* po = post_bf + ((long long)b * WH_ + m) * ND_;
    bf16_t* pt = probT + (long long)b * ND_ * WH_ + m;
    #pragma unroll
    for (int n = 0; n < ND_; ++n)
        pt[n * WH_] = (bf16_t)((float)po[n] * inv[n]);
    if (blockIdx.x == 0 && threadIdx.x < ND_)
        pi[b * ND_ + threadIdx.x] = colsum[b * ND_ + threadIdx.x] * (1.f / WH_);
}

// ---- graph adjacency: one block per batch ----------------------------------
// mean over nodes commutes with W_diag => dvec from row-sums of x2.
__global__ __launch_bounds__(256)
void graph_a_k(const float* __restrict__ x2f,
               const float* __restrict__ W_adj, const float* __restrict__ b_adj,
               const float* __restrict__ W_diag, const float* __restrict__ b_diag,
               float* __restrict__ Anorm)
{
    __shared__ float s_xa[DC_ * ND_];
    __shared__ float s_A[ND_ * ND_];
    __shared__ float s_dv[DC_];
    __shared__ float s_u[ND_];
    __shared__ float s_deg[ND_];
    __shared__ float s_rs[IN_];
    int b = blockIdx.x;
    const float* x2 = x2f + (long long)b * IN_ * ND_;
    int tid = threadIdx.x;

    {   // row sums of x2 over nodes
        float s = 0.f;
        const float* row = x2 + tid * ND_;
        for (int n = 0; n < ND_; ++n) s += row[n];
        s_rs[tid] = s;
    }
    __syncthreads();
    if (tid < DC_) {    // dvec = sigmoid(W_diag @ rowsum/64 + b_diag) - 0.5
        float a = 0.f;
        const float* wd = W_diag + tid * IN_;
        for (int c = 0; c < IN_; ++c) a += wd[c] * s_rs[c];
        a = a * (1.f / ND_) + b_diag[tid];
        s_dv[tid] = 1.f / (1.f + __expf(-a)) - 0.5f;
    }
    for (int idx = tid; idx < DC_ * ND_; idx += 256) {   // xa = W_adj@x2 + b
        int j = idx >> 6, n = idx & 63;
        float a = b_adj[j];
        const float* wa = W_adj + j * IN_;
        for (int c = 0; c < IN_; ++c) a += wa[c] * x2[c * ND_ + n];
        s_xa[idx] = a;
    }
    __syncthreads();
    if (tid < ND_) {                                     // u = colsum(xa)
        float s = 0.f;
        for (int j = 0; j < DC_; ++j) s += s_xa[j * ND_ + tid];
        s_u[tid] = s;
    }
    __syncthreads();
    for (int idx = tid; idx < ND_ * ND_; idx += 256) {   // A, relu, +I
        int n = idx >> 6, mm = idx & 63;
        float a = 0.5f * s_u[n] * s_u[mm];
        for (int j = 0; j < DC_; ++j) a += s_xa[j * ND_ + n] * s_dv[j] * s_xa[j * ND_ + mm];
        s_A[idx] = fmaxf(a, 0.f) + ((n == mm) ? 1.f : 0.f);
    }
    __syncthreads();
    if (tid < ND_) {                                     // deg = sum(A,1)^-1/2
        float s = 0.f;
        for (int n = 0; n < ND_; ++n) s += s_A[n * ND_ + tid];
        s_deg[tid] = rsqrtf(s);
    }
    __syncthreads();
    for (int idx = tid; idx < ND_ * ND_; idx += 256) {
        int n = idx >> 6, mm = idx & 63;
        Anorm[(long long)b * ND_ * ND_ + idx] = s_A[idx] * s_deg[n] * s_deg[mm];
    }
}

__global__ void graph_g1_k(const float* __restrict__ x2f, const float* __restrict__ Anorm,
                           float* __restrict__ g1)
{
    int i = blockIdx.x * 256 + threadIdx.x;   // B_*IN_*ND_
    int b = i >> 14;
    int rest = i & 16383;
    int c = rest >> 6, mm = rest & 63;
    const float* x2 = x2f + (long long)b * IN_ * ND_ + c * ND_;
    const float* An = Anorm + (long long)b * ND_ * ND_;
    float a = 0.f;
    for (int n = 0; n < ND_; ++n) a += x2[n] * An[n * ND_ + mm];
    g1[i] = a;
}

__global__ void graph_x2g_k(const float* __restrict__ g1, const float* __restrict__ gcn_w,
                            const float* __restrict__ x2f, bf16_t* __restrict__ x2g_bf)
{
    int i = blockIdx.x * 256 + threadIdx.x;
    int b = i >> 14;
    int rest = i & 16383;
    int o = rest >> 6, mm = rest & 63;
    const float* g1b = g1 + (long long)b * IN_ * ND_;
    const float* w = gcn_w + o * IN_;
    float a = 0.f;
    for (int ii = 0; ii < IN_; ++ii) a += w[ii] * g1b[ii * ND_ + mm];
    x2g_bf[i] = (bf16_t)(fmaxf(a, 0.f) + x2f[i]);
}

// ---- BN statistics over (batch, pixels) per channel ------------------------
__global__ __launch_bounds__(256)
void bn_stats_k(const float* __restrict__ Y, float* __restrict__ stat)
{
    int o = blockIdx.x;
    float s = 0.f, ss = 0.f;
    for (int i = threadIdx.x; i < B_ * WH_; i += 256) {
        int b = i >> 12, p = i & 4095;
        float v = Y[((long long)b * C_ + o) * WH_ + p];
        s += v; ss += v * v;
    }
    __shared__ float rs[256], rss[256];
    rs[threadIdx.x] = s; rss[threadIdx.x] = ss;
    __syncthreads();
    for (int st = 128; st > 0; st >>= 1) {
        if (threadIdx.x < st) { rs[threadIdx.x] += rs[threadIdx.x + st];
                                rss[threadIdx.x] += rss[threadIdx.x + st]; }
        __syncthreads();
    }
    if (threadIdx.x == 0) {
        float mean = rs[0] * (1.f / (B_ * WH_));
        float var  = rss[0] * (1.f / (B_ * WH_)) - mean * mean;
        stat[o * 2]     = mean;
        stat[o * 2 + 1] = rsqrtf(var + 1e-5f);
    }
}

__global__ void final_k(const float* __restrict__ y1, const float* __restrict__ y2,
                        const float* __restrict__ st1, const float* __restrict__ st2,
                        const float* __restrict__ gamma,  const float* __restrict__ beta,
                        const float* __restrict__ gamma2, const float* __restrict__ beta2,
                        const float* __restrict__ x, float* __restrict__ out)
{
    long long i = (long long)blockIdx.x * 256 + threadIdx.x;
    int o = (int)((i >> 12) & (C_ - 1));
    float xv = x[i];
    float a = (y1[i] - st1[o * 2]) * st1[o * 2 + 1] * gamma[o] + beta[o];
    out[i] = fmaxf(fmaxf(a, 0.f) + xv, 0.f);
    float b2 = (y2[i] - st2[o * 2]) * st2[o * 2 + 1] * gamma2[o] + beta2[o];
    out[(long long)B_ * C_ * WH_ + i] = fmaxf(fmaxf(b2, 0.f) + xv, 0.f);
}

// ---------------------------------------------------------------------------
extern "C" void kernel_launch(void* const* d_in, const int* in_sizes, int n_in,
                              void* d_out, int out_size, void* d_ws, size_t ws_size,
                              hipStream_t stream)
{
    (void)in_sizes; (void)n_in; (void)out_size; (void)ws_size;
    const float* x      = (const float*)d_in[0];
    const float* W_in   = (const float*)d_in[1];
    const float* b_in   = (const float*)d_in[2];
    const float* proto  = (const float*)d_in[3];
    const float* pi0    = (const float*)d_in[4];
    const float* W_adj  = (const float*)d_in[5];
    const float* b_adj  = (const float*)d_in[6];
    const float* W_diag = (const float*)d_in[7];
    const float* b_diag = (const float*)d_in[8];
    const float* gcn_w  = (const float*)d_in[9];
    const float* W_out  = (const float*)d_in[10];
    const float* b_out  = (const float*)d_in[11];
    const float* gamma  = (const float*)d_in[12];
    const float* beta   = (const float*)d_in[13];
    const float* W_out2 = (const float*)d_in[14];
    const float* b_out2 = (const float*)d_in[15];
    const float* gamma2 = (const float*)d_in[16];
    const float* beta2  = (const float*)d_in[17];
    float* out = (float*)d_out;

    char* ws = (char*)d_ws;
    size_t off = 0;
    auto alloc = [&](size_t bytes) -> void* {
        void* p = ws + off;
        off += bytes;
        off = (off + 255) & ~(size_t)255;
        return p;
    };
    bf16_t* xT    = (bf16_t*)alloc((size_t)B_ * WH_ * C_ * 2);
    bf16_t* Wi_b  = (bf16_t*)alloc((size_t)IN_ * C_ * 2);
    bf16_t* Wo_b  = (bf16_t*)alloc((size_t)C_ * IN_ * 2);
    bf16_t* Wo2_b = (bf16_t*)alloc((size_t)C_ * IN_ * 2);
    bf16_t* x1b   = (bf16_t*)alloc((size_t)B_ * IN_ * WH_ * 2);
    bf16_t* x1Tb  = (bf16_t*)alloc((size_t)B_ * WH_ * IN_ * 2);
    bf16_t* muT   = (bf16_t*)alloc((size_t)B_ * ND_ * IN_ * 2);
    float*  piB   = (float*) alloc((size_t)B_ * ND_ * 4);
    float*  likT  = (float*) alloc((size_t)B_ * ND_ * WH_ * 4);
    bf16_t* postb = (bf16_t*)alloc((size_t)B_ * WH_ * ND_ * 2);
    bf16_t* probT = (bf16_t*)alloc((size_t)B_ * ND_ * WH_ * 2);
    float*  colsum= (float*) alloc((size_t)B_ * ND_ * 4);
    float*  x2f   = (float*) alloc((size_t)B_ * IN_ * ND_ * 4);
    bf16_t* x2b   = (bf16_t*)alloc((size_t)B_ * IN_ * ND_ * 2);
    bf16_t* x2gb  = (bf16_t*)alloc((size_t)B_ * IN_ * ND_ * 2);
    float*  Anorm = (float*) alloc((size_t)B_ * ND_ * ND_ * 4);
    float*  g1    = (float*) alloc((size_t)B_ * IN_ * ND_ * 4);
    bf16_t* x4T   = (bf16_t*)alloc((size_t)B_ * WH_ * IN_ * 2);
    bf16_t* x3T   = (bf16_t*)alloc((size_t)B_ * WH_ * IN_ * 2);
    float*  y1    = (float*) alloc((size_t)B_ * C_ * WH_ * 4);
    float*  y2    = (float*) alloc((size_t)B_ * C_ * WH_ * 4);
    float*  st1   = (float*) alloc((size_t)C_ * 2 * 4);
    float*  st2   = (float*) alloc((size_t)C_ * 2 * 4);

    // prepare bf16 operands
    transpose_conv_k<<<dim3(WH_ / 32, C_ / 32, B_), 256, 0, stream>>>(x, xT);
    f2bf_k<<<(IN_ * C_) / 256, 256, 0, stream>>>(W_in,   Wi_b,  IN_ * C_);
    f2bf_k<<<(C_ * IN_) / 256, 256, 0, stream>>>(W_out,  Wo_b,  C_ * IN_);
    f2bf_k<<<(C_ * IN_) / 256, 256, 0, stream>>>(W_out2, Wo2_b, C_ * IN_);
    init_k<<<(B_ * ND_ * IN_) / 256, 256, 0, stream>>>(proto, pi0, muT, piB);

    // x1 = W_in @ x + b_in   (store channel-major and pixel-major bf16)
    gemm_bf16_k<<<dim3(WH_ / 64, IN_ / 64, B_), 128, 0, stream>>>(
        Wi_b, 0, C_, xT, (long long)WH_ * C_, C_, C_, b_in,
        nullptr, 0, 0,
        x1b,  (long long)IN_ * WH_, WH_,
        x1Tb, (long long)WH_ * IN_, IN_);

    for (int t = 0; t < 3; ++t) {
        // likT[n][m] = muT @ x1  (M=64,N=4096,K=256)
        gemm_bf16_k<<<dim3(WH_ / 64, ND_ / 64, B_), 128, 0, stream>>>(
            muT, (long long)ND_ * IN_, IN_, x1Tb, (long long)WH_ * IN_, IN_, IN_,
            nullptr,
            likT, (long long)ND_ * WH_, WH_,
            nullptr, 0, 0, nullptr, 0, 0);
        zero_k<<<1, 256, 0, stream>>>(colsum, B_ * ND_);
        em_post_k<<<dim3(WH_ / 256, B_), 256, 0, stream>>>(likT, piB, postb, colsum);
        em_prob_k<<<dim3(WH_ / 256, B_), 256, 0, stream>>>(postb, colsum, probT, piB);
        if (t < 2) {
            // muT[n][c] = probT @ x1^T  (M=64,N=256,K=4096)
            gemm_bf16_k<<<dim3(IN_ / 64, ND_ / 64, B_), 128, 0, stream>>>(
                probT, (long long)ND_ * WH_, WH_, x1b, (long long)IN_ * WH_, WH_, WH_,
                nullptr, nullptr, 0, 0,
                muT, (long long)ND_ * IN_, IN_,
                nullptr, 0, 0);
        }
    }

    // x2 = x1 @ prob  (M=256,N=64,K=4096) -> f32 + bf16
    gemm_bf16_k<<<dim3(ND_ / 64, IN_ / 64, B_), 128, 0, stream>>>(
        x1b, (long long)IN_ * WH_, WH_, probT, (long long)ND_ * WH_, WH_, WH_,
        nullptr,
        x2f, (long long)IN_ * ND_, ND_,
        x2b, (long long)IN_ * ND_, ND_,
        nullptr, 0, 0);

    // graph conv chain
    graph_a_k<<<B_, 256, 0, stream>>>(x2f, W_adj, b_adj, W_diag, b_diag, Anorm);
    graph_g1_k<<<(B_ * IN_ * ND_) / 256, 256, 0, stream>>>(x2f, Anorm, g1);
    graph_x2g_k<<<(B_ * IN_ * ND_) / 256, 256, 0, stream>>>(g1, gcn_w, x2f, x2gb);

    // x4T[m][c] = post @ x2^T  (M=4096,N=256,K=64)
    gemm_bf16_k<<<dim3(IN_ / 64, WH_ / 64, B_), 128, 0, stream>>>(
        postb, (long long)WH_ * ND_, ND_, x2b, (long long)IN_ * ND_, ND_, ND_,
        nullptr, nullptr, 0, 0,
        x4T, (long long)WH_ * IN_, IN_,
        nullptr, 0, 0);
    // x3T = post @ x2g^T
    gemm_bf16_k<<<dim3(IN_ / 64, WH_ / 64, B_), 128, 0, stream>>>(
        postb, (long long)WH_ * ND_, ND_, x2gb, (long long)IN_ * ND_, ND_, ND_,
        nullptr, nullptr, 0, 0,
        x3T, (long long)WH_ * IN_, IN_,
        nullptr, 0, 0);

    // y2 = W_out2 @ x4 + b_out2 ; y1 = W_out @ x3 + b_out  (M=1024,N=4096,K=256)
    gemm_bf16_k<<<dim3(WH_ / 64, C_ / 64, B_), 128, 0, stream>>>(
        Wo2_b, 0, IN_, x4T, (long long)WH_ * IN_, IN_, IN_, b_out2,
        y2, (long long)C_ * WH_, WH_,
        nullptr, 0, 0, nullptr, 0, 0);
    gemm_bf16_k<<<dim3(WH_ / 64, C_ / 64, B_), 128, 0, stream>>>(
        Wo_b, 0, IN_, x3T, (long long)WH_ * IN_, IN_, IN_, b_out,
        y1, (long long)C_ * WH_, WH_,
        nullptr, 0, 0, nullptr, 0, 0);

    bn_stats_k<<<C_, 256, 0, stream>>>(y1, st1);
    bn_stats_k<<<C_, 256, 0, stream>>>(y2, st2);
    final_k<<<(B_ * C_ * WH_) / 256, 256, 0, stream>>>(
        y1, y2, st1, st2, gamma, beta, gamma2, beta2, x, out);
}
